// FeatureComplete_52063593562699
// MI455X (gfx1250) — matrix operand order repaired and checked by reading, not compiled, not run on hardware
//
#include <hip/hip_runtime.h>
#include <hip/hip_bf16.h>
#include <math.h>

// ---------- types ----------
typedef __attribute__((ext_vector_type(16))) __bf16   v16bf;
typedef __attribute__((ext_vector_type(8)))  float    v8f;
typedef __attribute__((ext_vector_type(8)))  unsigned v8u;

#define NN      8192
#define NODE_D  128
#define HIDD    128
#define BB      4096
#define CTX1    32
#define SOURCE_ 4096

// hardware pack-convert: two f32 -> packed bf16 pair (low = a, high = b)
static __device__ __forceinline__ unsigned cvt_pk_bf16(float a, float b) {
    unsigned r;
    asm("v_cvt_pk_bf16_f32 %0, %1, %2" : "=v"(r) : "v"(a), "v"(b));
    return r;
}
// split one f32 pair into hi/lo bf16 packs (compensated bf16 representation)
static __device__ __forceinline__ void split_pair(float a, float b,
                                                  unsigned& hi, unsigned& lo) {
    hi = cvt_pk_bf16(a, b);
    const float ha = __uint_as_float(hi << 16);
    const float hb = __uint_as_float(hi & 0xffff0000u);
    lo = cvt_pk_bf16(a - ha, b - hb);
}

// ---------- zero the loss accumulator ----------
__global__ void k_zero(float* p) {
    if (threadIdx.x == 0 && blockIdx.x == 0) *p = 0.0f;
}

// ---------- generic row GEMM: Y[r,:] = E[r,:] @ W (+ scale*bias) ----------
// E: rows x 128, W: 128 x 128 row-major. One block per row, 128 threads.
__global__ __launch_bounds__(128) void k_rowgemm(const float* __restrict__ E,
                                                 const float* __restrict__ W,
                                                 const float* __restrict__ bias,
                                                 const float* __restrict__ para,
                                                 float* __restrict__ Y) {
    __shared__ float e[128];
    const int r = blockIdx.x, j = threadIdx.x;
    e[j] = E[(size_t)r * 128 + j];
    __syncthreads();
    float acc = 0.0f;
#pragma unroll 8
    for (int k = 0; k < 128; ++k) acc += e[k] * W[k * 128 + j];
    if (bias) {
        float bs = 1.0f;
        if (para) {  // simi path: bias scaled by mean(paraForCos)
            float sp = 0.0f;
#pragma unroll
            for (int k = 0; k < 8; ++k) sp += para[k];
            bs = sp * 0.125f;
        }
        acc += bs * bias[j];
    }
    Y[(size_t)r * 128 + j] = acc;
}

// ---------- pack B (8192x128 f32) into bf16 WMMA B fragments (contiguous k order) ----------
// Fragment (s = K-step of 32, t = 16-col tile, L = lane): 16 dwords contiguous,
// [0..7] = hi bf16 pairs, [8..15] = lo bf16 pairs.
// Lane L, dword v: K = s*32 + (L/16)*16 + 2v (+1), N = t*16 + (L%16).
__global__ __launch_bounds__(256) void k_pack(const float* __restrict__ X,
                                              unsigned* __restrict__ P) {
    const int tid = blockIdx.x * 256 + threadIdx.x;  // 524288 threads
    const int v = tid & 7;
    const int L = (tid >> 3) & 31;
    const int t = (tid >> 8) & 7;
    const int s = tid >> 11;
    const int k = s * 32 + (L >> 4) * 16 + v * 2;
    const int n = t * 16 + (L & 15);
    const float x0 = X[(size_t)k * 128 + n];
    const float x1 = X[(size_t)(k + 1) * 128 + n];
    unsigned hi, lo;
    split_pair(x0, x1, hi, lo);
    const size_t base = (size_t)(tid >> 3) * 16;  // fragment index * 16 dwords
    P[base + v]     = hi;
    P[base + 8 + v] = lo;
}

// ---------- big GEMM: OUT = relu(A @ X + b), A: 8192x8192 f32 streamed ----------
// Split-bf16 compensation: A = Ahi + Alo (in-register via v_cvt_pk_bf16_f32),
// X pre-split hi/lo. acc += Ahi*Bhi + Alo*Bhi + Ahi*Blo (error ~2^-16).
// Block 256 = 8 waves. Workgroup tile: 64 rows x 128 cols.
// Wave w: rows [bx*64 + (w%4)*16, +16), cols [(w/4)*64, +64) -> 4 accumulators.
__global__ __launch_bounds__(256) void k_big(const float* __restrict__ A0,
                                             const float* __restrict__ A1,
                                             const unsigned* __restrict__ P0,
                                             const unsigned* __restrict__ P1,
                                             const float* __restrict__ b0,
                                             const float* __restrict__ b1,
                                             float* __restrict__ O0,
                                             float* __restrict__ O1) {
    const float* A; const unsigned* P; const float* bias; float* O;
    if (blockIdx.y == 0) { A = A0; P = P0; bias = b0; O = O0; }
    else                 { A = A1; P = P1; bias = b1; O = O1; }

    const int tid  = threadIdx.x;
    const int w    = tid >> 5;
    const int L    = tid & 31;
    const int mSub = w & 3;   // which 16-row band
    const int nH   = w >> 2;  // which 64-col half
    const int half = L >> 4;
    const int m0   = blockIdx.x * 64 + mSub * 16;
    const float* Arow = A + (size_t)(m0 + (L & 15)) * NN;

    v8f acc[4] = {};

    for (int s = 0; s < 256; ++s) {
        // ---- build A hi/lo fragments from streamed f32 (hw pack-convert) ----
        v8u ahiU, aloU;
#pragma unroll
        for (int g = 0; g < 2; ++g) {
            const int base = s * 32 + half * 16 + g * 8;
            const float4 f0 = *(const float4*)(Arow + base);
            const float4 f1 = *(const float4*)(Arow + base + 4);
            const float fv[8] = {f0.x, f0.y, f0.z, f0.w, f1.x, f1.y, f1.z, f1.w};
#pragma unroll
            for (int q = 0; q < 4; ++q) {
                unsigned hi, lo;
                split_pair(fv[2 * q], fv[2 * q + 1], hi, lo);
                ahiU[g * 4 + q] = hi;
                aloU[g * 4 + q] = lo;
            }
        }
        const v16bf ahi = __builtin_bit_cast(v16bf, ahiU);
        const v16bf alo = __builtin_bit_cast(v16bf, aloU);

        if (s + 8 < 256)  // prefetch the streamed row 8 K-steps ahead
            __builtin_prefetch(Arow + (s + 8) * 32 + half * 8, 0, 3);

        // ---- 4 N-tiles x 3 compensated WMMAs ----
#pragma unroll
        for (int j = 0; j < 4; ++j) {
            const int t = nH * 4 + j;
            const v8u* bp = (const v8u*)(P + (((size_t)s * 8 + t) * 32 + L) * 16);
            const v16bf bhi = __builtin_bit_cast(v16bf, bp[0]);
            const v16bf blo = __builtin_bit_cast(v16bf, bp[1]);
            acc[j] = __builtin_amdgcn_wmma_f32_16x16x32_bf16(
                false, ahi, false, bhi, (short)0, acc[j], false, false);
            acc[j] = __builtin_amdgcn_wmma_f32_16x16x32_bf16(
                false, alo, false, bhi, (short)0, acc[j], false, false);
            acc[j] = __builtin_amdgcn_wmma_f32_16x16x32_bf16(
                false, ahi, false, blo, (short)0, acc[j], false, false);
        }
    }

    // ---- epilogue: bias + relu, C layout VGPR r -> M = r + 8*half ----
#pragma unroll
    for (int j = 0; j < 4; ++j) {
        const int col = nH * 64 + j * 16 + (L & 15);
        const float bv = bias[col];
#pragma unroll
        for (int r = 0; r < 8; ++r) {
            const int row = m0 + r + half * 8;
            const float v = acc[j][r] + bv;
            O[(size_t)row * 128 + col] = v > 0.0f ? v : 0.0f;
        }
    }
}

// ---------- normalize trainfeature[:, :64] per node ----------
__global__ __launch_bounds__(256) void k_norm(const float* __restrict__ tf,
                                              float* __restrict__ tfn) {
    const int w = threadIdx.x >> 5, L = threadIdx.x & 31;
    const int n = blockIdx.x * 8 + w;
    const float v0 = tf[(size_t)n * 128 + L];
    const float v1 = tf[(size_t)n * 128 + 32 + L];
    float s = v0 * v0 + v1 * v1;
#pragma unroll
    for (int m = 16; m >= 1; m >>= 1) s += __shfl_xor(s, m, 32);
    const float inv = 1.0f / (sqrtf(s) + 1e-8f);
    tfn[(size_t)n * 64 + L]      = v0 * inv;
    tfn[(size_t)n * 64 + 32 + L] = v1 * inv;
}

// ---------- weighted average over TOPK for simi path ----------
__global__ __launch_bounds__(128) void k_wavg(const float* __restrict__ bsnf,
                                              const float* __restrict__ para,
                                              float* __restrict__ y) {
    const int b = blockIdx.x, d = threadIdx.x;
    float acc = 0.0f;
#pragma unroll
    for (int k = 0; k < 8; ++k) acc += para[k] * bsnf[((size_t)b * 8 + k) * 128 + d];
    y[(size_t)b * 128 + d] = acc * 0.125f;
}

// ---------- attention: one wave per batch row, lane = context position ----------
__global__ __launch_bounds__(128) void k_attn(const int* __restrict__ node_rd,
                                              const float* __restrict__ Q,
                                              const float* __restrict__ K,
                                              const float* __restrict__ V,
                                              const float* __restrict__ tfn,
                                              float* __restrict__ att) {
    const int w = threadIdx.x >> 5, c = threadIdx.x & 31;
    const int b = blockIdx.x * 4 + w;
    const int ic = node_rd[b * CTX1 + c] + (c == 0 ? SOURCE_ : 0);
    const int i0 = __shfl(ic, 0, 32);

    // cosine similarity over normalized first-64 features
    float cosv = 0.0f;
    for (int j = 0; j < 64; ++j)
        cosv += tfn[(size_t)i0 * 64 + j] * tfn[(size_t)ic * 64 + j];

    float a[8];
#pragma unroll
    for (int h = 0; h < 8; ++h) {
        float sc = 0.0f;
#pragma unroll
        for (int d = 0; d < 16; ++d)
            sc += Q[(size_t)i0 * 128 + h * 16 + d] * K[(size_t)ic * 128 + h * 16 + d];
        sc = sc * 0.25f + cosv;  // 1/sqrt(16)
        float mx = sc;
#pragma unroll
        for (int m = 16; m >= 1; m >>= 1) mx = fmaxf(mx, __shfl_xor(mx, m, 32));
        const float e = expf(sc - mx);
        float sm = e;
#pragma unroll
        for (int m = 16; m >= 1; m >>= 1) sm += __shfl_xor(sm, m, 32);
        a[h] = e / sm;
    }

#pragma unroll
    for (int h = 0; h < 8; ++h) {
        float p[16];
#pragma unroll
        for (int d = 0; d < 16; ++d)
            p[d] = a[h] * V[(size_t)ic * 128 + h * 16 + d];
#pragma unroll
        for (int d = 0; d < 16; ++d)
#pragma unroll
            for (int m = 16; m >= 1; m >>= 1) p[d] += __shfl_xor(p[d], m, 32);
        if (c == 0) {
#pragma unroll
            for (int d = 0; d < 16; ++d) {
                const float x = p[d];
                att[(size_t)b * 128 + h * 16 + d] = x > 0.0f ? x : (expf(x) - 1.0f);
            }
        }
    }
}

// ---------- final: result = [emb_g | emb_att | simi] @ lin_W + lin_b, loss ----------
__global__ __launch_bounds__(128) void k_final(const float* __restrict__ embg,
                                               const float* __restrict__ att,
                                               const float* __restrict__ simi,
                                               const int* __restrict__ bidx,
                                               const float* __restrict__ W,
                                               const float* __restrict__ bias,
                                               const float* __restrict__ target,
                                               float* __restrict__ out,
                                               float* __restrict__ loss) {
    __shared__ float f[384];
    const int b = blockIdx.x, j = threadIdx.x;
    const int g = bidx[b];
    f[j]       = embg[(size_t)g * 128 + j];
    f[128 + j] = att[(size_t)b * 128 + j];
    f[256 + j] = simi[(size_t)b * 128 + j];
    __syncthreads();
    float acc = bias[j];
#pragma unroll 8
    for (int k = 0; k < 384; ++k) acc += f[k] * W[k * 128 + j];
    out[(size_t)b * 128 + j] = acc;
    if (j == 5) {
        const float d = target[(size_t)b * 128 + 5] - acc;
        atomicAdd(loss, d * d * (1.0f / (float)BB));
    }
}

extern "C" void kernel_launch(void* const* d_in, const int* in_sizes, int n_in,
                              void* d_out, int out_size, void* d_ws, size_t ws_size,
                              hipStream_t stream) {
    // inputs in setup_inputs() dict order
    const float* adj        = (const float*)d_in[0];
    const float* graph      = (const float*)d_in[1];
    const float* node_emb   = (const float*)d_in[2];
    const float* trainfeat  = (const float*)d_in[3];
    const float* target     = (const float*)d_in[4];
    const float* bsnf       = (const float*)d_in[5];
    // d_in[6] node_emb_rd, d_in[7] feature: unused (zeros)
    const int*   bidx       = (const int*)d_in[8];
    const int*   node_rd    = (const int*)d_in[9];
    const float* trans_W    = (const float*)d_in[10];
    const float* trans_b    = (const float*)d_in[11];
    const float* para       = (const float*)d_in[12];
    const float* gcn_W      = (const float*)d_in[13];
    const float* gcn_b      = (const float*)d_in[14];
    const float* gcnE_W     = (const float*)d_in[15];
    const float* gcnE_b     = (const float*)d_in[16];
    const float* Wq         = (const float*)d_in[17];
    const float* Wk         = (const float*)d_in[18];
    const float* Wv         = (const float*)d_in[19];
    const float* lin_W      = (const float*)d_in[20];
    const float* lin_b      = (const float*)d_in[21];

    // workspace layout
    float* ws = (float*)d_ws;
    float*    X1  = ws;                           // 8192x128
    float*    X2  = X1 + 1048576;
    unsigned* P1  = (unsigned*)(X2 + 1048576);    // packed bf16 hi/lo fragments
    unsigned* P2  = P1 + 1048576;
    float*    EG  = (float*)(P2 + 1048576);       // emb_gcn
    float*    EGE = EG + 1048576;                 // emb_gcnE
    float*    Qm  = EGE + 1048576;
    float*    Km  = Qm + 1048576;
    float*    Vm  = Km + 1048576;
    float*    TFN = Vm + 1048576;                 // 8192x64
    float*    Yw  = TFN + 524288;                 // 4096x128
    float*    SIM = Yw + 524288;
    float*    ATT = SIM + 524288;

    float* result = (float*)d_out;
    float* loss   = result + (size_t)BB * 128;

    k_zero<<<1, 32, 0, stream>>>(loss);

    // X1 = node_emb_gcn @ gcn_W ; X2 = trainfeature @ gcnE_W
    k_rowgemm<<<NN, 128, 0, stream>>>(node_emb, gcn_W, nullptr, nullptr, X1);
    k_rowgemm<<<NN, 128, 0, stream>>>(trainfeat, gcnE_W, nullptr, nullptr, X2);

    // split-pack B-side into WMMA fragment layout
    k_pack<<<2048, 256, 0, stream>>>(X1, P1);
    k_pack<<<2048, 256, 0, stream>>>(X2, P2);

    // emb_gcn = relu(adj @ X1 + b); emb_gcnE = relu(graph @ X2 + b)  [WMMA]
    dim3 gb(128, 2);
    k_big<<<gb, 256, 0, stream>>>(adj, graph, P1, P2, gcn_b, gcnE_b, EG, EGE);

    // project ALL nodes once; gather later (16x fewer FLOPs than gathered GEMM)
    k_rowgemm<<<NN, 128, 0, stream>>>(EGE, Wq, nullptr, nullptr, Qm);
    k_rowgemm<<<NN, 128, 0, stream>>>(EGE, Wk, nullptr, nullptr, Km);
    k_rowgemm<<<NN, 128, 0, stream>>>(EGE, Wv, nullptr, nullptr, Vm);

    k_norm<<<NN / 8, 256, 0, stream>>>(trainfeat, TFN);

    // simi = ((1/8) sum_k para_k * bsnf_k) @ trans_W + mean(para)*trans_b
    k_wavg<<<BB, 128, 0, stream>>>(bsnf, para, Yw);
    k_rowgemm<<<BB, 128, 0, stream>>>(Yw, trans_W, trans_b, para, SIM);

    k_attn<<<BB / 4, 128, 0, stream>>>(node_rd, Qm, Km, Vm, TFN, ATT);

    k_final<<<BB, 128, 0, stream>>>(EG, ATT, SIM, bidx, lin_W, lin_b, target,
                                    result, loss);
}
